// DeformConv_38852274159751
// MI455X (gfx1250) — compile-verified, matrix operand order checked by reading, not run on hardware
//
#include <hip/hip_runtime.h>
#include <math.h>

// DCNv2 for MI455X (gfx1250): offset-conv -> bilinear sample (im2col in LDS,
// bf16) -> WMMA GEMM (v_wmma_f32_16x16x32_bf16) -> folded BN + ReLU.
// Workspace: [0, 7962624) = om (B x 27 x H x W f32)
//            [8MiB, 8MiB+1179648) = w_dcn packed to bf16 A-fragments
// Requires ws_size >= 9,568,256 bytes.

#define KK 9
#define PAD 1
#define EPS 1e-5f

static const int Bn  = 8;
static const int Cc  = 256;
static const int Hh  = 96;
static const int Ww  = 96;
static const int Oo  = 256;
static const int HW  = Hh * Ww;       // 9216
static const int CHW = Cc * HW;       // 2359296
static const int KTOT = Cc * KK;      // 2304
static const int KC   = 32;           // bf16 WMMA K per step
static const int NKC  = KTOT / KC;    // 72 K-chunks
static const int NOT  = Oo / 16;      // 16 o-tiles
static const int NPIX = Bn * HW;      // 73728
static const int PIXT = 32;           // pixels per block (2 N-tiles)
static const int NPT  = NPIX / PIXT;  // 2304 blocks (tiles stay inside a row)

typedef __attribute__((ext_vector_type(16))) __bf16 v16bf;
typedef __attribute__((ext_vector_type(8)))  float  v8f;

__device__ __forceinline__ unsigned short f2bf(float f) {
  unsigned u = __float_as_uint(f);
  u = u + 0x7FFFu + ((u >> 16) & 1u);   // round-to-nearest-even
  return (unsigned short)(u >> 16);
}

// 16-bit A/B 16x32 WMMA fragment: (lane, element) -> K index  (ISA 7.12.2)
__device__ __forceinline__ int frag_k(int lane, int e) {
  return ((e < 8) ? e : (e + 8)) + ((lane >> 4) & 1) * 8;
}

// ---------------------------------------------------------------------------
// Kernel 1: offset conv (27 out channels, 3x3, C=256). 2 pixels per thread,
// weight chunks staged in LDS (broadcast reads), raw om -> workspace.
// ---------------------------------------------------------------------------
__global__ __launch_bounds__(256) void offset_conv_kernel(
    const float* __restrict__ x, const float* __restrict__ w_off,
    const float* __restrict__ b_off, float* __restrict__ om)
{
  __shared__ float lds_w[32 * KK * 27];   // 31104 B

  int tid = threadIdx.x;
  int p0 = blockIdx.x * 512 + tid;
  int p1 = p0 + 256;

  float acc0[27], acc1[27];
  #pragma unroll
  for (int i = 0; i < 27; i++) { acc0[i] = 0.f; acc1[i] = 0.f; }

  int b0 = p0 / HW, r0 = p0 % HW, h0 = r0 / Ww, w0 = r0 % Ww;
  int b1 = p1 / HW, r1 = p1 % HW, h1 = r1 / Ww, w1 = r1 % Ww;
  int xb0 = b0 * CHW, xb1 = b1 * CHW;

  int off0[9], off1[9];
  bool ok0[9], ok1[9];
  #pragma unroll
  for (int t = 0; t < 9; t++) {
    int ky = t / 3 - PAD, kx = t % 3 - PAD;
    int yy0 = h0 + ky, xx0 = w0 + kx;
    int yy1 = h1 + ky, xx1 = w1 + kx;
    ok0[t] = (yy0 >= 0) && (yy0 < Hh) && (xx0 >= 0) && (xx0 < Ww);
    ok1[t] = (yy1 >= 0) && (yy1 < Hh) && (xx1 >= 0) && (xx1 < Ww);
    off0[t] = yy0 * Ww + xx0;
    off1[t] = yy1 * Ww + xx1;
  }

  for (int cc = 0; cc < Cc; cc += 32) {
    __syncthreads();
    for (int j = tid; j < 32 * KK * 27; j += 256) {
      int oc = j % 27, rest = j / 27, tap = rest % KK, ci = rest / KK;
      lds_w[j] = w_off[(oc * Cc + (cc + ci)) * KK + tap];
    }
    __syncthreads();
    for (int ci = 0; ci < 32; ci++) {
      int plane0 = xb0 + (cc + ci) * HW;
      int plane1 = xb1 + (cc + ci) * HW;
      #pragma unroll
      for (int t = 0; t < 9; t++) {
        float xv0 = ok0[t] ? x[plane0 + off0[t]] : 0.f;
        float xv1 = ok1[t] ? x[plane1 + off1[t]] : 0.f;
        const float* wp = &lds_w[(ci * KK + t) * 27];
        #pragma unroll
        for (int oc = 0; oc < 27; oc++) {
          float wv = wp[oc];
          acc0[oc] += xv0 * wv;
          acc1[oc] += xv1 * wv;
        }
      }
    }
  }

  #pragma unroll
  for (int oc = 0; oc < 27; oc++) {
    float bb = b_off[oc];
    om[b0 * 27 * HW + oc * HW + r0] = acc0[oc] + bb;
    om[b1 * 27 * HW + oc * HW + r1] = acc1[oc] + bb;
  }
}

// ---------------------------------------------------------------------------
// Kernel 2: pack w_dcn (O x C x 3 x 3 f32) into bf16 WMMA A-fragment order:
// wsA[((ot*72 + kc)*32 + lane)*16 + e], one thread per bf16 element.
// ---------------------------------------------------------------------------
__global__ __launch_bounds__(256) void pack_wdcn_kernel(
    const float* __restrict__ w_dcn, unsigned short* __restrict__ wsA)
{
  int idx  = blockIdx.x * 256 + threadIdx.x;   // < 16*72*32*16 = 589824
  int e    = idx & 15;
  int lane = (idx >> 4) & 31;
  int t    = idx >> 9;
  int kc   = t % NKC;
  int ot   = t / NKC;
  int M    = lane & 15;
  int o    = ot * 16 + M;
  int K    = frag_k(lane, e);
  int g    = kc * KC + K;
  int c    = g / KK, tap = g % KK;
  wsA[idx] = f2bf(w_dcn[(o * Cc + c) * KK + tap]);
}

// ---------------------------------------------------------------------------
// Kernel 3: fused sample + GEMM + BN + ReLU.
// Block = 32 pixels x 256 output channels; 8 waves x 2 o-tiles x 2 N-tiles.
// Ping-pong B buffers -> 1 barrier per K-step, 4 WMMAs per barrier.
// ---------------------------------------------------------------------------
__global__ __launch_bounds__(256) void dcn_wmma_kernel(
    const float* __restrict__ x, const float* __restrict__ om,
    const unsigned short* __restrict__ wsA,
    const float* __restrict__ b_dcn, const float* __restrict__ gamma,
    const float* __restrict__ beta, const float* __restrict__ mean,
    const float* __restrict__ var, float* __restrict__ out)
{
  __shared__ int   la00[288], la01[288], la10[288], la11[288];  // y*W+x, clamped
  __shared__ float lw00[288], lw01[288], lw10[288], lw11[288];  // bilinear*ok*mask
  __shared__ int   lxbase[PIXT], lprem[PIXT];
  __shared__ float lscale[256], lbias[256];
  __shared__ __align__(32) unsigned short ldsB[2][2][512];      // [pp][nt] 32x16 bf16

  int tid  = threadIdx.x;
  int tile = blockIdx.x;

  // Fold BN + conv bias into per-channel scale/bias:
  // out = (acc + b_dcn - mean)*gamma*rsqrt(var+eps) + beta = acc*sc + bi
  {
    int o = tid;
    float sc = gamma[o] * rsqrtf(var[o] + EPS);
    lscale[o] = sc;
    lbias[o]  = beta[o] + (b_dcn[o] - mean[o]) * sc;
  }
  if (tid < PIXT) {
    int p = tile * PIXT + tid;
    int b = p / HW, rem = p % HW;
    lxbase[tid] = b * CHW;       // C == O, also valid as output batch base
    lprem[tid]  = rem;           // tiles never cross a row (W%32==0)
  }
  // Per (pixel n, tap): bilinear corner offsets + weights (x ok x mask)
  for (int i = tid; i < PIXT * KK; i += 256) {
    int n = i / KK, tap = i % KK;
    int p = tile * PIXT + n;
    int b = p / HW, rem = p % HW, h = rem / Ww, w = rem % Ww;
    int base = b * 27 * HW + rem;
    float offy = om[base + (2 * tap) * HW];
    float offx = om[base + (2 * tap + 1) * HW];
    float mraw = om[base + (18 + tap) * HW];
    float m = 1.f / (1.f + __expf(-mraw));
    int ky = tap / 3, kx = tap % 3;
    float py = offy + (float)(h + ky - PAD);
    float px = offx + (float)(w + kx - PAD);
    float fy = floorf(py), fx = floorf(px);
    int y0 = (int)fy, x0 = (int)fx;
    float wy = py - fy, wx = px - fx;
    int y1 = y0 + 1, x1 = x0 + 1;
    float oy0 = (y0 >= 0 && y0 < Hh) ? 1.f : 0.f;
    float oy1 = (y1 >= 0 && y1 < Hh) ? 1.f : 0.f;
    float ox0 = (x0 >= 0 && x0 < Ww) ? 1.f : 0.f;
    float ox1 = (x1 >= 0 && x1 < Ww) ? 1.f : 0.f;
    lw00[i] = (1.f - wy) * (1.f - wx) * oy0 * ox0 * m;
    lw01[i] = (1.f - wy) * wx * oy0 * ox1 * m;
    lw10[i] = wy * (1.f - wx) * oy1 * ox0 * m;
    lw11[i] = wy * wx * oy1 * ox1 * m;
    int cy0 = min(max(y0, 0), Hh - 1) * Ww;
    int cy1 = min(max(y1, 0), Hh - 1) * Ww;
    int cx0 = min(max(x0, 0), Ww - 1);
    int cx1 = min(max(x1, 0), Ww - 1);
    la00[i] = cy0 + cx0;
    la01[i] = cy0 + cx1;
    la10[i] = cy1 + cx0;
    la11[i] = cy1 + cx1;
  }
  __syncthreads();

  int lane = tid & 31;
  int wv   = tid >> 5;            // wave 0..7
  int ot0  = wv * 2, ot1 = wv * 2 + 1;

  // sampling role: thread fills B entries for K=Kl, pixels n0,n0+8,n0+16,n0+24
  int Kl = tid & 31;
  int n0 = tid >> 5;
  int lslot = ((Kl >> 3) & 1) * 16;             // B-fragment lane half
  int eS = (Kl & 7) + ((Kl >> 4) & 1) * 8;      // B-fragment element

  int tap = Kl % KK;              // K-row position, updated incrementally
  int plane_c = (Kl / KK) * HW;   // c*HW, updated incrementally (g += 32/step)

  v8f acc00 = {}, acc01 = {}, acc10 = {}, acc11 = {};
  const v16bf* Aglob = reinterpret_cast<const v16bf*>(wsA);

  for (int kc = 0; kc < NKC; kc++) {
    int pp = kc & 1;
    #pragma unroll
    for (int q = 0; q < 4; q++) {
      int n  = n0 + q * 8;
      int gi = n * KK + tap;
      int base = lxbase[n] + plane_c;
      float v = lw00[gi] * x[base + la00[gi]]
              + lw01[gi] * x[base + la01[gi]]
              + lw10[gi] * x[base + la10[gi]]
              + lw11[gi] * x[base + la11[gi]];
      ldsB[pp][n >> 4][((n & 15) + lslot) * 16 + eS] = f2bf(v);
    }
    __syncthreads();
    // Safe single-barrier ping-pong: each thread's WMMA waits (DScnt) on its
    // B-fragment loads, so buffer pp is fully consumed before that thread can
    // reach the next barrier; rewriting pp two iterations later is race-free.
    v16bf b0 = *reinterpret_cast<const v16bf*>(&ldsB[pp][0][lane * 16]);
    v16bf b1 = *reinterpret_cast<const v16bf*>(&ldsB[pp][1][lane * 16]);
    v16bf a0 = Aglob[(ot0 * NKC + kc) * 32 + lane];
    v16bf a1 = Aglob[(ot1 * NKC + kc) * 32 + lane];
    acc00 = __builtin_amdgcn_wmma_f32_16x16x32_bf16(
        false, a0, false, b0, (short)0, acc00, false, false);
    acc01 = __builtin_amdgcn_wmma_f32_16x16x32_bf16(
        false, a0, false, b1, (short)0, acc01, false, false);
    acc10 = __builtin_amdgcn_wmma_f32_16x16x32_bf16(
        false, a1, false, b0, (short)0, acc10, false, false);
    acc11 = __builtin_amdgcn_wmma_f32_16x16x32_bf16(
        false, a1, false, b1, (short)0, acc11, false, false);
    // advance K by 32: tap += 32 mod 9 (=5), carry into channel plane
    tap += 5;
    int ov = (tap >= KK) ? 1 : 0;
    tap -= ov * KK;
    plane_c += (3 + ov) * HW;
  }

  // Epilogue: C/D layout -> lanes 0-15: M=r, lanes 16-31: M=r+8; N = lane%16
  int Nn  = lane & 15;
  int hiL = lane >> 4;
  int pb0 = lxbase[Nn],      pr0 = lprem[Nn];        // N-tile 0
  int pb1 = lxbase[16 + Nn], pr1 = lprem[16 + Nn];   // N-tile 1
  #pragma unroll
  for (int r = 0; r < 8; r++) {
    int oA = ot0 * 16 + r + hiL * 8;
    int oB = ot1 * 16 + r + hiL * 8;
    float sA = lscale[oA], bA = lbias[oA];
    float sB = lscale[oB], bB = lbias[oB];
    out[pb0 + oA * HW + pr0] = fmaxf(acc00[r] * sA + bA, 0.f);
    out[pb1 + oA * HW + pr1] = fmaxf(acc01[r] * sA + bA, 0.f);
    out[pb0 + oB * HW + pr0] = fmaxf(acc10[r] * sB + bB, 0.f);
    out[pb1 + oB * HW + pr1] = fmaxf(acc11[r] * sB + bB, 0.f);
  }
}

// ---------------------------------------------------------------------------
extern "C" void kernel_launch(void* const* d_in, const int* in_sizes, int n_in,
                              void* d_out, int out_size, void* d_ws, size_t ws_size,
                              hipStream_t stream)
{
  (void)in_sizes; (void)n_in; (void)out_size; (void)ws_size;
  const float* x     = (const float*)d_in[0];
  const float* w_off = (const float*)d_in[1];
  const float* b_off = (const float*)d_in[2];
  const float* w_dcn = (const float*)d_in[3];
  const float* b_dcn = (const float*)d_in[4];
  const float* gamma = (const float*)d_in[5];
  const float* beta  = (const float*)d_in[6];
  const float* rmean = (const float*)d_in[7];
  const float* rvar  = (const float*)d_in[8];
  float* out = (float*)d_out;

  float* om = (float*)d_ws;                                        // 7,962,624 B
  unsigned short* wsA =
      (unsigned short*)((char*)d_ws + (size_t)(8u << 20));         // 1,179,648 B

  offset_conv_kernel<<<NPIX / 512, 256, 0, stream>>>(x, w_off, b_off, om);
  pack_wdcn_kernel<<<(NOT * NKC * 32 * 16) / 256, 256, 0, stream>>>(w_dcn, wsA);
  dcn_wmma_kernel<<<NPT, 256, 0, stream>>>(x, om, wsA, b_dcn, gamma, beta,
                                           rmean, rvar, out);
}
